// GraphAttentionLayer_17265768530255
// MI455X (gfx1250) — matrix operand order repaired and checked by reading, not compiled, not run on hardware
//
#include <hip/hip_runtime.h>
#include <hip/hip_bf16.h>

typedef __attribute__((ext_vector_type(16))) __bf16 v16bf;
typedef __attribute__((ext_vector_type(8)))  __bf16 v8bf;
typedef __attribute__((ext_vector_type(8)))  float  v8f;
typedef __attribute__((ext_vector_type(4)))  float  v4f;

#define N_NODES 8192
#define IN_F    256
#define OUT_F   64
#define HEADS   4
#define ALPHA   0.2f
#define LOG2E   1.4426950408889634f

// ---------------------------------------------------------------------------
// Kernel 1: convert x -> bf16 (row major), W -> W^T bf16 (WT[h][o][f])
// ---------------------------------------------------------------------------
__global__ void prep_kernel(const float* __restrict__ x,
                            const float* __restrict__ W,
                            __bf16* __restrict__ xbf,
                            __bf16* __restrict__ WTbf) {
    int i = blockIdx.x * blockDim.x + threadIdx.x;
    if (i < N_NODES * IN_F) xbf[i] = (__bf16)x[i];
    if (i < HEADS * IN_F * OUT_F) {
        int h   = i / (IN_F * OUT_F);
        int rem = i - h * (IN_F * OUT_F);
        int f   = rem / OUT_F;
        int o   = rem - f * OUT_F;
        WTbf[(h * OUT_F + o) * IN_F + f] = (__bf16)W[i];
    }
}

// ---------------------------------------------------------------------------
// Kernel 2: h = x @ W per head (bf16 WMMA).  One wave = 16 rows x 64 cols.
// Writes h as f32 row-major (for s1/s2) and hT as bf16 (WMMA B operand).
// ---------------------------------------------------------------------------
__global__ void __launch_bounds__(256)
hgemm_kernel(const __bf16* __restrict__ xbf,
             const __bf16* __restrict__ WTbf,
             float*  __restrict__ hF,
             __bf16* __restrict__ hTbf) {
    const int lane  = threadIdx.x & 31;
    const int wave  = threadIdx.x >> 5;
    const int strip = blockIdx.x * (blockDim.x >> 5) + wave;   // 0..2047
    const int head  = strip >> 9;
    const int pbase = (strip & 511) * 16;
    const int m     = lane & 15;
    const int hi    = lane >> 4;
    const int c0    = hi * 16;    // A-tile K sub-offset per half-wave
    const int khalf = hi * 16;    // B-tile K half per half-wave

    v8f acc[4] = {};
    const __bf16* xrow = xbf + (size_t)(pbase + m) * IN_F;

    for (int kb = 0; kb < IN_F; kb += 32) {
        // A tile: x rows (16x32), layout: elems 0..15 = K c0..c0+15
        v8bf alo = *(const v8bf*)(xrow + kb + c0);
        v8bf ahi = *(const v8bf*)(xrow + kb + c0 + 8);
        v16bf A;
        #pragma unroll
        for (int j = 0; j < 8; ++j) { A[j] = alo[j]; A[j + 8] = ahi[j]; }
        #pragma unroll
        for (int t = 0; t < 4; ++t) {
            // B tile: W^T row o = t*16+m, contiguous 16 K values
            const __bf16* wrow = WTbf + ((size_t)(head * OUT_F + t * 16 + m)) * IN_F
                                      + kb + khalf;
            v16bf B = *(const v16bf*)wrow;
            acc[t] = __builtin_amdgcn_wmma_f32_16x16x32_bf16(
                         false, A, false, B, (short)0, acc[t], false, false);
        }
    }
    #pragma unroll
    for (int t = 0; t < 4; ++t) {
        const int col = t * 16 + m;
        #pragma unroll
        for (int r = 0; r < 8; ++r) {
            const int row = r + hi * 8;          // C/D layout: VGPR r -> M = r or r+8
            float v = acc[t][r];
            hF  [(((size_t)head * N_NODES) + pbase + row) * OUT_F + col] = v;
            hTbf[(((size_t)head * OUT_F) + col) * N_NODES + pbase + row] = (__bf16)v;
        }
    }
}

// ---------------------------------------------------------------------------
// Kernel 3: s1 = (h . a1) * log2(e), s2 = (h . a2) * log2(e)
// Pre-scaling by log2(e) lets the hot kernel use a bare v_exp_f32 (exp2).
// ---------------------------------------------------------------------------
__global__ void s_kernel(const float* __restrict__ hF,
                         const float* __restrict__ a,
                         float* __restrict__ s1,
                         float* __restrict__ s2) {
    int idx = blockIdx.x * blockDim.x + threadIdx.x;
    if (idx >= HEADS * N_NODES) return;
    int head = idx / N_NODES;
    const float* hrow = hF + (size_t)idx * OUT_F;
    const float* a1 = a + head * 2 * OUT_F;
    const float* a2 = a1 + OUT_F;
    float acc1 = 0.f, acc2 = 0.f;
    #pragma unroll 8
    for (int o = 0; o < OUT_F; ++o) {
        float hv = hrow[o];
        acc1 += hv * a1[o];
        acc2 += hv * a2[o];
    }
    s1[idx] = acc1 * LOG2E;
    s2[idx] = acc2 * LOG2E;
}

// ---------------------------------------------------------------------------
// Kernel 4: fused masked softmax-attention x h.  One wave = (head, 16 rows).
// - streams adj exactly once (256 MB -> ~11us at 23.3 TB/s)
// - exp weights: adj in {0,1} so mask = single mul; lrelu = max(t, 0.2t);
//   s pre-scaled by log2e so exp is a bare v_exp_f32
// - Z via a 5th WMMA against an all-ones B tile: zacc[r] = Z[row of acc[t][r]]
//   replicated in every lane -> normalization needs no cross-lane traffic
// - double-buffered loads: next chunk's adj/s1/B issued before current chunk's
//   VALU + 5 WMMAs, hiding memory latency at low occupancy
// ---------------------------------------------------------------------------
__global__ void __launch_bounds__(256)
gat_kernel(const float* __restrict__ adj,
           const __bf16* __restrict__ hTbf,
           const float* __restrict__ s1,
           const float* __restrict__ s2,
           float* __restrict__ out) {
    const int lane  = threadIdx.x & 31;
    const int wave  = threadIdx.x >> 5;
    const int strip = blockIdx.x * (blockDim.x >> 5) + wave;   // 0..2047
    const int head  = strip >> 9;
    const int pbase = (strip & 511) * 16;
    const int m     = lane & 15;
    const int hi    = lane >> 4;
    const int c0    = hi * 16;
    const int khalf = hi * 16;

    const float  s2v  = s2[head * N_NODES + pbase + m];      // pre-scaled by log2e
    const float* arow = adj + (size_t)(pbase + m) * N_NODES;
    const float* s1h  = s1 + head * N_NODES;
    const __bf16* hTh = hTbf + (size_t)head * OUT_F * N_NODES;

    v8f acc[4] = {};
    v8f zacc   = {};

    v16bf Bones;
    #pragma unroll
    for (int i = 0; i < 16; ++i) Bones[i] = (__bf16)1.0f;

    // ---- preload chunk 0 ----
    v4f cad0, cad1, cad2, cad3, csv0, csv1, csv2, csv3;
    v16bf cB0, cB1, cB2, cB3;
    {
        const float* ap = arow + c0;
        const float* sp = s1h + c0;
        cad0 = *(const v4f*)(ap);      cad1 = *(const v4f*)(ap + 4);
        cad2 = *(const v4f*)(ap + 8); cad3 = *(const v4f*)(ap + 12);
        csv0 = *(const v4f*)(sp);      csv1 = *(const v4f*)(sp + 4);
        csv2 = *(const v4f*)(sp + 8); csv3 = *(const v4f*)(sp + 12);
        cB0 = *(const v16bf*)(hTh + (size_t)(0  + m) * N_NODES + khalf);
        cB1 = *(const v16bf*)(hTh + (size_t)(16 + m) * N_NODES + khalf);
        cB2 = *(const v16bf*)(hTh + (size_t)(32 + m) * N_NODES + khalf);
        cB3 = *(const v16bf*)(hTh + (size_t)(48 + m) * N_NODES + khalf);
    }

    for (int qb = 0; qb < N_NODES; qb += 32) {
        // ---- issue next chunk's loads (wrap on last iter; values unused) ----
        const int qn = (qb + 32) & (N_NODES - 1);
        const float* ap = arow + qn + c0;
        const float* sp = s1h + qn + c0;
        v4f nad0 = *(const v4f*)(ap);      v4f nad1 = *(const v4f*)(ap + 4);
        v4f nad2 = *(const v4f*)(ap + 8); v4f nad3 = *(const v4f*)(ap + 12);
        v4f nsv0 = *(const v4f*)(sp);      v4f nsv1 = *(const v4f*)(sp + 4);
        v4f nsv2 = *(const v4f*)(sp + 8); v4f nsv3 = *(const v4f*)(sp + 12);
        v16bf nB0 = *(const v16bf*)(hTh + (size_t)(0  + m) * N_NODES + qn + khalf);
        v16bf nB1 = *(const v16bf*)(hTh + (size_t)(16 + m) * N_NODES + qn + khalf);
        v16bf nB2 = *(const v16bf*)(hTh + (size_t)(32 + m) * N_NODES + qn + khalf);
        v16bf nB3 = *(const v16bf*)(hTh + (size_t)(48 + m) * N_NODES + qn + khalf);
        __builtin_prefetch(arow + ((qb + 64) & (N_NODES - 1)) + c0, 0, 1);

        // ---- build exp-weight A tile from current chunk (k = 16*hi + i order, same as B) ----
        float av[16], sv[16];
        #pragma unroll
        for (int j = 0; j < 4; ++j) {
            av[j]      = cad0[j]; av[j + 4]  = cad1[j];
            av[j + 8]  = cad2[j]; av[j + 12] = cad3[j];
            sv[j]      = csv0[j]; sv[j + 4]  = csv1[j];
            sv[j + 8]  = csv2[j]; sv[j + 12] = csv3[j];
        }
        v16bf A;
        #pragma unroll
        for (int i = 0; i < 16; ++i) {
            float t = s2v + sv[i];               // (s2+s1)*log2e
            float e = fmaxf(t, ALPHA * t);       // leaky_relu (alpha<1)
            float w = av[i] * exp2f(e);          // adj in {0,1}: mask == multiply
            A[i] = (__bf16)w;
        }

        // ---- 4 output-tile WMMAs + 1 partition-sum WMMA ----
        acc[0] = __builtin_amdgcn_wmma_f32_16x16x32_bf16(false, A, false, cB0,
                                                         (short)0, acc[0], false, false);
        acc[1] = __builtin_amdgcn_wmma_f32_16x16x32_bf16(false, A, false, cB1,
                                                         (short)0, acc[1], false, false);
        acc[2] = __builtin_amdgcn_wmma_f32_16x16x32_bf16(false, A, false, cB2,
                                                         (short)0, acc[2], false, false);
        acc[3] = __builtin_amdgcn_wmma_f32_16x16x32_bf16(false, A, false, cB3,
                                                         (short)0, acc[3], false, false);
        zacc   = __builtin_amdgcn_wmma_f32_16x16x32_bf16(false, A, false, Bones,
                                                         (short)0, zacc,   false, false);

        // ---- rotate pipeline ----
        cad0 = nad0; cad1 = nad1; cad2 = nad2; cad3 = nad3;
        csv0 = nsv0; csv1 = nsv1; csv2 = nsv2; csv3 = nsv3;
        cB0 = nB0; cB1 = nB1; cB2 = nB2; cB3 = nB3;
    }

    // ---- normalize and store: zacc[r] already holds Z for row (r + hi*8) ----
    float zinv[8];
    #pragma unroll
    for (int r = 0; r < 8; ++r) zinv[r] = 1.0f / zacc[r];

    #pragma unroll
    for (int t4 = 0; t4 < 4; ++t4) {
        const int col = head * OUT_F + t4 * 16 + m;
        #pragma unroll
        for (int r = 0; r < 8; ++r) {
            const int row = r + hi * 8;
            out[(size_t)(pbase + row) * (HEADS * OUT_F) + col] = acc[t4][r] * zinv[r];
        }
    }
}

// ---------------------------------------------------------------------------
extern "C" void kernel_launch(void* const* d_in, const int* in_sizes, int n_in,
                              void* d_out, int out_size, void* d_ws, size_t ws_size,
                              hipStream_t stream) {
    const float* x   = (const float*)d_in[0];   // (8192, 256)
    const float* adj = (const float*)d_in[1];   // (8192, 8192)
    const float* W   = (const float*)d_in[2];   // (4, 256, 64)
    const float* a   = (const float*)d_in[3];   // (4, 128, 1)
    float* out = (float*)d_out;                 // (8192, 256)

    char* ws = (char*)d_ws;
    __bf16* xbf  = (__bf16*)(ws);                        //  4,194,304 B
    __bf16* WTbf = (__bf16*)(ws + 4194304);              //    131,072 B
    float*  hF   = (float*) (ws + 4325376);              //  8,388,608 B
    __bf16* hTbf = (__bf16*)(ws + 12713984);             //  4,194,304 B
    float*  s1   = (float*) (ws + 16908288);             //    131,072 B
    float*  s2   = (float*) (ws + 17039360);             //    131,072 B (end 17,170,432)

    // 1) precision conversion / transposes
    prep_kernel<<<(N_NODES * IN_F + 255) / 256, 256, 0, stream>>>(x, W, xbf, WTbf);

    // 2) h = x @ W per head: 2048 wave-strips, 8 waves/block
    hgemm_kernel<<<256, 256, 0, stream>>>(xbf, WTbf, hF, hTbf);

    // 3) attention logits s1, s2 (pre-scaled by log2e)
    s_kernel<<<(HEADS * N_NODES + 255) / 256, 256, 0, stream>>>(hF, a, s1, s2);

    // 4) fused mask + softmax + att@h, single pass over adj
    gat_kernel<<<256, 256, 0, stream>>>(adj, hTbf, s1, s2, out);
}
